// Head_58445914964306
// MI455X (gfx1250) — matrix operand order repaired and checked by reading, not compiled, not run on hardware
//
#include <hip/hip_runtime.h>
#include <hip/hip_bf16.h>

typedef __attribute__((ext_vector_type(16))) _Float16 v16h;
typedef __attribute__((ext_vector_type(8)))  _Float16 v8h;
typedef __attribute__((ext_vector_type(8)))  float    v8f;
typedef __attribute__((ext_vector_type(4)))  float    v4f;

static constexpr int SEQ  = 64;    // T
static constexpr int CDIM = 768;   // embed
static constexpr int HD   = 64;    // head size
static constexpr int KC   = CDIM / 32;  // 24 K-chunks for projections
static constexpr int XS_STRIDE  = 40;   // padded halves per staged-x row (conflict-free b128)
static constexpr int ROW_STRIDE = 72;   // padded halves per row of k,q,vT,P
static constexpr int SF_STRIDE  = 68;   // padded floats per score row

// LDS arena layout (bytes):
//  Region A (phase-overlayed):
//    phase 1: xs0 @0 (5120), xs1 @5120 (5120)
//    phase 2: Sf  @0 (17408), redm @17408 (1024), reds @18432 (1024)
//  Region B (persistent): kbuf, qbuf, vTb, Pb  (4 x 9216)
static constexpr int REGION_A = 19456;
static constexpr int SMEM_BYTES = REGION_A + 4 * (SEQ * ROW_STRIDE * 2);

// Build a v16h fragment from two 16-byte LDS/global chunks.
__device__ inline v16h frag2(const _Float16* p0, const _Float16* p1) {
  v16h r;
  ((v8h*)&r)[0] = *(const v8h*)p0;
  ((v8h*)&r)[1] = *(const v8h*)p1;
  return r;
}

__device__ inline v8f wmma_f16(v16h a, v16h b, v8f c) {
  return __builtin_amdgcn_wmma_f32_16x16x32_f16(false, a, false, b, (short)0, c, false, false);
}

// ---------------------------------------------------------------------------
// Prep: pack weights (fp32 [768][64]) into f16 WMMA fragment layouts in d_ws.
// Slot f = matht (0..11): mat0/1 = Wk^T,Wq^T in A-layout, mat2 = Wv in B-layout.
// Flat: (((matht*24 + kc)*32 + lane)*16 + j) halves.
// A-layout (row h = ht*16 + lane&15): j<8 -> c = kc*32 + (lane>>4)*8 + j
//                                     j>=8 -> c = kc*32 + 16 + (lane>>4)*8 + (j-8)
// B-layout (col h = ht*16 + lane&15): c = kc*32 + (lane>>4)*16 + j
// ---------------------------------------------------------------------------
__global__ void pack_weights(const float* __restrict__ Wk, const float* __restrict__ Wq,
                             const float* __restrict__ Wv, _Float16* __restrict__ ws) {
  int idx = blockIdx.x * 256 + threadIdx.x;       // one lane-task = 16 halves
  if (idx >= 3 * 4 * 24 * 32) return;
  int lane = idx & 31;
  int kc   = (idx >> 5) % 24;
  int ht   = ((idx >> 5) / 24) & 3;
  int mat  = idx / (32 * 24 * 4);
  const float* W = (mat == 0) ? Wk : (mat == 1) ? Wq : Wv;
  _Float16* dst = ws + (size_t)idx * 16;
  int col = lane & 15, hi = lane >> 4;
  int h = ht * 16 + col;
  if (mat < 2) {
    #pragma unroll
    for (int j = 0; j < 16; ++j) {
      int c = kc * 32 + ((j < 8) ? (hi * 8 + j) : (16 + hi * 8 + (j - 8)));
      dst[j] = (_Float16)W[c * HD + h];
    }
  } else {
    #pragma unroll
    for (int j = 0; j < 16; ++j) {
      int c = kc * 32 + hi * 16 + j;
      dst[j] = (_Float16)W[c * HD + h];
    }
  }
}

// ---------------------------------------------------------------------------
// Main: one workgroup (256 threads = 8 waves) per batch element.
// ---------------------------------------------------------------------------
__global__ __launch_bounds__(256) void attn_head(const float* __restrict__ x,
                                                 const _Float16* __restrict__ wp,
                                                 float* __restrict__ out) {
  __shared__ __align__(16) unsigned char smem[SMEM_BYTES];
  float*    Sf   = (float*)smem;                       // 64 x 68 f32 (overlays xs)
  float*    redm = (float*)(smem + 17408);             // 256 f32
  float*    reds = (float*)(smem + 18432);             // 256 f32
  _Float16* kbuf = (_Float16*)(smem + REGION_A);       // k[t][h]
  _Float16* qbuf = kbuf + SEQ * ROW_STRIDE;            // q[t][h] (pre-scaled)
  _Float16* vTb  = qbuf + SEQ * ROW_STRIDE;            // v^T[h][t]
  _Float16* Pb   = vTb + HD * ROW_STRIDE;              // probs f16

  const int b    = blockIdx.x;
  const int tid  = threadIdx.x;
  const int w    = tid >> 5;
  const int lane = tid & 31;
  const int col  = lane & 15;
  const int hi   = lane >> 4;
  const int tt   = w & 3;        // fixed 16-wide tile along T (or H for v path)
  const int ht0  = w >> 2;       // 0..1
  const int ht1  = ht0 + 2;      // 2..3

  const float* xb = x + (size_t)b * (SEQ * CDIM);

  v8f accK0 = {}, accK1 = {}, accQ0 = {}, accQ1 = {}, accV0 = {}, accV1 = {};

  const int st_row  = tid >> 2;        // 0..63
  const int st_part = (tid & 3) * 8;   // 0,8,16,24

  // ================= Phase 1: projections (software-pipelined) =============
  v4f f0, f1;
  {
    const float* src = xb + st_row * CDIM + st_part;
    f0 = *(const v4f*)(src);
    f1 = *(const v4f*)(src + 4);
  }
  for (int kc = 0; kc < KC; ++kc) {
    // double-buffered staging area (computed, not a pointer array: lld chokes
    // on static initializers holding LDS addrspacecasts)
    _Float16* xs = (_Float16*)(smem + (kc & 1) * 5120);
    // convert prefetched chunk kc and store to LDS
    v8h hs;
    #pragma unroll
    for (int j = 0; j < 4; ++j) { hs[j] = (_Float16)f0[j]; hs[4 + j] = (_Float16)f1[j]; }
    *(v8h*)(xs + st_row * XS_STRIDE + st_part) = hs;
    // issue global load for chunk kc+1 (consumed next iteration)
    if (kc + 1 < KC) {
      const float* src = xb + st_row * CDIM + (kc + 1) * 32 + st_part;
      f0 = *(const v4f*)(src);
      f1 = *(const v4f*)(src + 4);
      if (kc + 2 < KC) __builtin_prefetch(src + 32, 0, 3);
    }
    __syncthreads();

    // B-frag of x^T (k,q path): col t = tt*16+col, elems {hi*8+0..7, 16+hi*8+0..7}
    const _Float16* bxp = xs + (tt * 16 + col) * XS_STRIDE + hi * 8;
    v16h bx = frag2(bxp, bxp + 16);
    // A-frag of x (v path): row t = tt*16+col, K run = hi*16 + 0..15
    const _Float16* axp = xs + (tt * 16 + col) * XS_STRIDE + hi * 16;
    v16h ax = frag2(axp, axp + 8);

    // pre-packed weight frags (L2/WGP$-resident, contiguous 32B per lane)
    #define WFRAG(matht) (*(const v16h*)(wp + (((size_t)(matht) * KC + kc) * 32 + lane) * 16))
    v16h aK0 = WFRAG(0 * 4 + ht0);
    v16h aK1 = WFRAG(0 * 4 + ht1);
    v16h aQ0 = WFRAG(1 * 4 + ht0);
    v16h aQ1 = WFRAG(1 * 4 + ht1);
    v16h bV0 = WFRAG(2 * 4 + ht0);
    v16h bV1 = WFRAG(2 * 4 + ht1);
    #undef WFRAG

    accK0 = wmma_f16(aK0, bx, accK0);   // D = k^T tile (m=h, n=t)
    accK1 = wmma_f16(aK1, bx, accK1);
    accQ0 = wmma_f16(aQ0, bx, accQ0);
    accQ1 = wmma_f16(aQ1, bx, accQ1);
    accV0 = wmma_f16(ax, bV0, accV0);   // D = v tile (m=t, n=h)
    accV1 = wmma_f16(ax, bV1, accV1);
    // no trailing barrier: next iteration stores to the other xs buffer;
    // the next __syncthreads orders reuse of this buffer two iterations out.
  }
  __syncthreads();

  // store projections to LDS; D frag (m = 8*hi + i, n = col) -> contiguous b128
  {
    const float qscale = 0.125f;   // 1/sqrt(64) folded into q
    v8h h0, h1, h2, h3, h4, h5;
    #pragma unroll
    for (int i = 0; i < 8; ++i) {
      h0[i] = (_Float16)accK0[i];            h1[i] = (_Float16)accK1[i];
      h2[i] = (_Float16)(accQ0[i] * qscale); h3[i] = (_Float16)(accQ1[i] * qscale);
      h4[i] = (_Float16)accV0[i];            h5[i] = (_Float16)accV1[i];
    }
    int tg = tt * 16 + col;                       // n-index of k/q tiles = t
    *(v8h*)(kbuf + tg * ROW_STRIDE + ht0 * 16 + hi * 8) = h0;
    *(v8h*)(kbuf + tg * ROW_STRIDE + ht1 * 16 + hi * 8) = h1;
    *(v8h*)(qbuf + tg * ROW_STRIDE + ht0 * 16 + hi * 8) = h2;
    *(v8h*)(qbuf + tg * ROW_STRIDE + ht1 * 16 + hi * 8) = h3;
    int hg0 = ht0 * 16 + col, hg1 = ht1 * 16 + col;   // n-index of v tiles = h
    *(v8h*)(vTb + hg0 * ROW_STRIDE + tt * 16 + hi * 8) = h4;
    *(v8h*)(vTb + hg1 * ROW_STRIDE + tt * 16 + hi * 8) = h5;
  }
  __syncthreads();

  // ================= Phase 2: S = (q_scaled) k^T, causal ===================
  #pragma unroll
  for (int pick = 0; pick < 2; ++pick) {
    int mt = pick ? ht1 : ht0;
    if (tt > mt) continue;                 // tile fully above diagonal: skip
    v8f s = {};
    #pragma unroll
    for (int kc2 = 0; kc2 < 2; ++kc2) {
      const _Float16* ap = qbuf + (mt * 16 + col) * ROW_STRIDE + kc2 * 32 + hi * 8;
      v16h aq = frag2(ap, ap + 16);
      const _Float16* bp = kbuf + (tt * 16 + col) * ROW_STRIDE + kc2 * 32 + hi * 8;
      v16h bk = frag2(bp, bp + 16);
      s = wmma_f16(aq, bk, s);
    }
    int sg = tt * 16 + col;
    #pragma unroll
    for (int i = 0; i < 8; ++i) {
      int tg = mt * 16 + hi * 8 + i;
      Sf[tg * SF_STRIDE + sg] = s[i];
    }
  }
  __syncthreads();

  // ============ softmax: 4 threads per row, 16 cols each ===================
  {
    const int r    = tid >> 2;
    const int c0   = (tid & 3) * 16;
    const float* row = Sf + r * SF_STRIDE;
    v4f q0 = *(const v4f*)(row + c0);
    v4f q1 = *(const v4f*)(row + c0 + 4);
    v4f q2 = *(const v4f*)(row + c0 + 8);
    v4f q3 = *(const v4f*)(row + c0 + 12);
    float vals[16];
    #pragma unroll
    for (int j = 0; j < 4; ++j) {
      vals[j] = q0[j]; vals[4 + j] = q1[j]; vals[8 + j] = q2[j]; vals[12 + j] = q3[j];
    }
    float m = -3.0e38f;
    #pragma unroll
    for (int j = 0; j < 16; ++j) if (c0 + j <= r) m = fmaxf(m, vals[j]);
    redm[tid] = m;
    __syncthreads();
    float mx = fmaxf(fmaxf(redm[r * 4 + 0], redm[r * 4 + 1]),
                     fmaxf(redm[r * 4 + 2], redm[r * 4 + 3]));
    float e[16];
    float s = 0.f;
    #pragma unroll
    for (int j = 0; j < 16; ++j) {
      float v = (c0 + j <= r) ? __expf(vals[j] - mx) : 0.f;
      e[j] = v; s += v;
    }
    reds[tid] = s;
    __syncthreads();
    float sum = (reds[r * 4 + 0] + reds[r * 4 + 1]) + (reds[r * 4 + 2] + reds[r * 4 + 3]);
    float inv = 1.f / sum;
    v8h p0, p1;
    #pragma unroll
    for (int j = 0; j < 8; ++j) { p0[j] = (_Float16)(e[j] * inv); p1[j] = (_Float16)(e[8 + j] * inv); }
    *(v8h*)(Pb + r * ROW_STRIDE + c0)     = p0;
    *(v8h*)(Pb + r * ROW_STRIDE + c0 + 8) = p1;
  }
  __syncthreads();

  // ================= Phase 3: O = P v ======================================
  float* ob = out + (size_t)b * (SEQ * HD);
  #pragma unroll
  for (int pick = 0; pick < 2; ++pick) {
    int mt = pick ? ht1 : ht0;
    v8f o = {};
    #pragma unroll
    for (int kc3 = 0; kc3 < 2; ++kc3) {
      const _Float16* ap = Pb + (mt * 16 + col) * ROW_STRIDE + kc3 * 32 + hi * 8;
      v16h aP = frag2(ap, ap + 16);
      const _Float16* bp = vTb + (tt * 16 + col) * ROW_STRIDE + kc3 * 32 + hi * 8;
      v16h bv = frag2(bp, bp + 16);
      o = wmma_f16(aP, bv, o);
    }
    int hg = tt * 16 + col;
    #pragma unroll
    for (int i = 0; i < 8; ++i) {
      int tg = mt * 16 + hi * 8 + i;
      ob[tg * HD + hg] = o[i];
    }
  }
}

extern "C" void kernel_launch(void* const* d_in, const int* in_sizes, int n_in,
                              void* d_out, int out_size, void* d_ws, size_t ws_size,
                              hipStream_t stream) {
  const float* x  = (const float*)d_in[0];
  const float* Wk = (const float*)d_in[1];
  const float* Wq = (const float*)d_in[2];
  const float* Wv = (const float*)d_in[3];
  float* out = (float*)d_out;
  _Float16* wpack = (_Float16*)d_ws;   // needs 3*768*64*2 = 294,912 bytes

  // pack weights into WMMA fragment layouts (9216 lane-tasks)
  pack_weights<<<36, 256, 0, stream>>>(Wk, Wq, Wv, wpack);
  // one workgroup per batch element
  attn_head<<<2048, 256, 0, stream>>>(x, wpack, out);
}
